// EuclideanCodebook_87162066305133
// MI455X (gfx1250) — compile-verified, matrix operand order checked by reading, not run on hardware
//
#include <hip/hip_runtime.h>

// CDNA5 / gfx1250: wave32, v_wmma_f32_16x16x32_bf16, async global->LDS staging.
typedef __attribute__((ext_vector_type(16))) __bf16 v16bf;
typedef __attribute__((ext_vector_type(8)))  __bf16 v8bf;
typedef __attribute__((ext_vector_type(8)))  float  v8f;
typedef int v4i_vs __attribute__((vector_size(16)));  // matches builtin param

#define AS1 __attribute__((address_space(1)))  // global
#define AS3 __attribute__((address_space(3)))  // LDS

#define DIM 128            // vector dimension (GEMM K)
#define ROWS_PER_WAVE 16   // M tile
#define WAVES_PER_BLOCK 8  // 256 threads
#define CHUNK 64           // codes staged into LDS per buffer
#define BSTRIDE (DIM + 8)  // padded LDS row stride (272B): kills bank conflicts

#if __has_builtin(__builtin_amdgcn_global_load_async_to_lds_b128) && \
    __has_builtin(__builtin_amdgcn_global_load_async_to_lds_b32)
#define HAS_ASYNC 1
#else
#define HAS_ASYNC 0
#endif

#if HAS_ASYNC
#if __has_builtin(__builtin_amdgcn_s_wait_asynccnt)
#define WAIT_ASYNC() __builtin_amdgcn_s_wait_asynccnt(0)
#else
#define WAIT_ASYNC() asm volatile("s_wait_asynccnt 0x0" ::: "memory")
#endif
#else
#define WAIT_ASYNC() ((void)0)  // fallback staging uses ds_store; barrier covers it
#endif

// round-to-nearest-even f32 -> bf16
__device__ __forceinline__ __bf16 f2bf(float f) {
    union { float f; unsigned u; } in; in.f = f;
    unsigned r = (in.u + 0x7FFFu + ((in.u >> 16) & 1u)) >> 16;
    union { unsigned short s; __bf16 b; } out; out.s = (unsigned short)r;
    return out.b;
}

// ---------------------------------------------------------------------------
// Prep kernel (once, tiny): embed f32 -> bf16 table + ||e||^2 per code.
// ---------------------------------------------------------------------------
__global__ __launch_bounds__(256) void
prep_codes(const float* __restrict__ embed, __bf16* __restrict__ ebf,
           float* __restrict__ e2, int K) {
    const int code = blockIdx.x * blockDim.x + threadIdx.x;
    if (code >= K) return;
    const float* src = embed + (size_t)code * DIM;
    v8bf* dst = (v8bf*)(ebf + (size_t)code * DIM);
    float ss = 0.0f;
#pragma unroll
    for (int i = 0; i < DIM / 8; ++i) {
        const float4 f0 = *(const float4*)(src + i * 8);
        const float4 f1 = *(const float4*)(src + i * 8 + 4);
        ss += f0.x * f0.x + f0.y * f0.y + f0.z * f0.z + f0.w * f0.w;
        ss += f1.x * f1.x + f1.y * f1.y + f1.z * f1.z + f1.w * f1.w;
        v8bf o;
        o[0] = f2bf(f0.x); o[1] = f2bf(f0.y); o[2] = f2bf(f0.z); o[3] = f2bf(f0.w);
        o[4] = f2bf(f1.x); o[5] = f2bf(f1.y); o[6] = f2bf(f1.z); o[7] = f2bf(f1.w);
        dst[i] = o;
    }
    e2[code] = ss;
}

// A tile: 16 rows of x, converted once per wave into the WMMA A layout:
//   lane = half*16 + m, slice s: elems 0..7  <- K = s*32 + half*8 + {0..7}
//                                elems 8..15 <- K = s*32 + 16 + half*8 + {0..7}
__device__ __forceinline__ void load_a(const float* __restrict__ p, int half,
                                       v16bf v[4]) {
#pragma unroll
    for (int s = 0; s < 4; ++s) {
        const float* r0 = p + s * 32 + half * 8;
        const float* r1 = r0 + 16;
        float f[16];
        *(float4*)(f + 0)  = *(const float4*)(r0 + 0);
        *(float4*)(f + 4)  = *(const float4*)(r0 + 4);
        *(float4*)(f + 8)  = *(const float4*)(r1 + 0);
        *(float4*)(f + 12) = *(const float4*)(r1 + 4);
#pragma unroll
        for (int i = 0; i < 16; ++i) v[s][i] = f2bf(f[i]);
    }
}

// Stage one chunk (64 codes x 256B + 64 e2 floats) into LDS.
// 256 threads: each thread moves 64B (4 x async b128) of one code row.
__device__ __forceinline__ void stage_chunk(const __bf16* __restrict__ ebf,
                                            const float* __restrict__ e2arr,
                                            __bf16* lbt, float* le2,
                                            int c0, int tid) {
    const int code = tid >> 2;
    const int seg  = (tid & 3) * 4;  // 4 consecutive 16B segments
    const __bf16* g = ebf + (size_t)(c0 + code) * DIM + seg * 8;
    __bf16* l = lbt + code * BSTRIDE + seg * 8;
#if HAS_ASYNC
#pragma unroll
    for (int k = 0; k < 4; ++k)
        __builtin_amdgcn_global_load_async_to_lds_b128(
            (AS1 v4i_vs*)(g + k * 8), (AS3 v4i_vs*)(l + k * 8), 0, 0);
    if (tid < CHUNK)
        __builtin_amdgcn_global_load_async_to_lds_b32(
            (AS1 int*)(e2arr + c0 + tid), (AS3 int*)(le2 + tid), 0, 0);
#else
#pragma unroll
    for (int k = 0; k < 4; ++k)
        *(uint4*)(l + k * 8) = *(const uint4*)(g + k * 8);
    if (tid < CHUNK) le2[tid] = e2arr[c0 + tid];
#endif
}

// ---------------------------------------------------------------------------
// Main kernel: block = 8 waves x 16 rows. B tiles staged once per BLOCK into
// LDS (async, double-buffered), read back as ds_load_b128 WMMA operands.
// ---------------------------------------------------------------------------
__global__ __launch_bounds__(256) void
vq_argmax_gather(const float* __restrict__ x, const float* __restrict__ embed,
                 const __bf16* __restrict__ ebf, const float* __restrict__ e2arr,
                 float* __restrict__ outq, float* __restrict__ outi, int nCodes) {
    __shared__ __align__(16) __bf16 bt[2][CHUNK * BSTRIDE];
    __shared__ float e2t[2][CHUNK];

    const int tid  = threadIdx.x;
    const int lane = tid & 31;
    const int wave = tid >> 5;
    const int half = lane >> 4;
    const int r    = lane & 15;
    const int row0 = (blockIdx.x * WAVES_PER_BLOCK + wave) * ROWS_PER_WAVE;

    stage_chunk(ebf, e2arr, bt[0], e2t[0], 0, tid);  // prime buffer 0

    v16bf a[4];
    load_a(x + (size_t)(row0 + r) * DIM, half, a);

    float best[8];
    int   bidx[8];
#pragma unroll
    for (int v = 0; v < 8; ++v) { best[v] = -3.4e38f; bidx[v] = 0; }

    const int nChunks = nCodes / CHUNK;
    for (int c = 0; c < nChunks; ++c) {
        WAIT_ASYNC();        // this wave's stagings have landed in LDS
        __syncthreads();     // everyone's have; prev compute on other buf done

        if (c + 1 < nChunks)
            stage_chunk(ebf, e2arr, bt[(c + 1) & 1], e2t[(c + 1) & 1],
                        (c + 1) * CHUNK, tid);

        const __bf16* base = bt[c & 1];
        const float*  e2b  = e2t[c & 1];

#pragma unroll
        for (int sub = 0; sub < 4; ++sub) {
            const int colL = sub * 16 + r;
            const int col  = c * CHUNK + colL;
            const __bf16* bp = base + colL * BSTRIDE;

            v16bf b[4];
#pragma unroll
            for (int s = 0; s < 4; ++s) {
                const v8bf lo = *(const v8bf*)(bp + s * 32 + half * 8);
                const v8bf hi = *(const v8bf*)(bp + s * 32 + 16 + half * 8);
                b[s] = __builtin_shufflevector(lo, hi, 0, 1, 2, 3, 4, 5, 6, 7,
                                               8, 9, 10, 11, 12, 13, 14, 15);
            }
            const float e2 = e2b[colL];

            v8f acc = {};
#pragma unroll
            for (int s = 0; s < 4; ++s)
                acc = __builtin_amdgcn_wmma_f32_16x16x32_bf16(
                    false, a[s], false, b[s], (short)0, acc, false, false);

#pragma unroll
            for (int v = 0; v < 8; ++v) {
                const float d = fmaf(2.0f, acc[v], -e2);
                const bool g = d > best[v];
                best[v] = g ? d : best[v];
                bidx[v] = g ? col : bidx[v];
            }
        }
    }

    // reduce argmax across the 16 columns held by each half-wave
#pragma unroll
    for (int off = 1; off < 16; off <<= 1) {
#pragma unroll
        for (int v = 0; v < 8; ++v) {
            const float ov = __shfl_xor(best[v], off, 32);
            const int   oi = __shfl_xor(bidx[v], off, 32);
            const bool g = (ov > best[v]) || (ov == best[v] && oi < bidx[v]);
            best[v] = g ? ov : best[v];
            bidx[v] = g ? oi : bidx[v];
        }
    }

    // C layout: lanes 0-15 slot v -> row v ; lanes 16-31 slot v -> row v+8
    if (r == 0) {
#pragma unroll
        for (int v = 0; v < 8; ++v)
            outi[row0 + half * 8 + v] = (float)bidx[v];
    }

    // gather: quantize[row] = embed[best]; 32 lanes x float4 = 512B per row
    const float4* __restrict__ e4 = (const float4*)embed;
    float4* __restrict__ o4 = (float4*)outq;
#pragma unroll
    for (int row = 0; row < ROWS_PER_WAVE; ++row) {
        const int idx = __shfl(bidx[row & 7], (row >> 3) * 16, 32);
        o4[(size_t)(row0 + row) * (DIM / 4) + lane] =
            e4[(size_t)idx * (DIM / 4) + lane];
    }
}

extern "C" void kernel_launch(void* const* d_in, const int* in_sizes, int n_in,
                              void* d_out, int out_size, void* d_ws, size_t ws_size,
                              hipStream_t stream) {
    const float* x     = (const float*)d_in[0];   // [16,4096,128] f32
    const float* embed = (const float*)d_in[1];   // [1024,128]    f32
    const int N = in_sizes[0] / DIM;              // 65536 rows
    const int K = in_sizes[1] / DIM;              // 1024 codes

    float* outq = (float*)d_out;                  // quantize [N,128] f32
    float* outi = outq + (size_t)N * DIM;         // embed_ind [N] (as f32)

    // workspace: bf16 code table (K*DIM*2 B) + e2 (K*4 B) => ~260 KB
    __bf16* ebf = (__bf16*)d_ws;
    float*  e2  = (float*)((char*)d_ws + (size_t)K * DIM * sizeof(__bf16));

    prep_codes<<<(K + 255) / 256, 256, 0, stream>>>(embed, ebf, e2, K);

    const int rowsPerBlock = ROWS_PER_WAVE * WAVES_PER_BLOCK;  // 128
    vq_argmax_gather<<<N / rowsPerBlock, 256, 0, stream>>>(
        x, embed, ebf, e2, outq, outi, K);
}
